// GCN_CNN_14302241096267
// MI455X (gfx1250) — compile-verified, matrix operand order checked by reading, not run on hardware
//
#include <hip/hip_runtime.h>

// ---------------- CDNA5 WMMA types ----------------
typedef __bf16 v16bf __attribute__((ext_vector_type(16)));
typedef __bf16 v8bf  __attribute__((ext_vector_type(8)));
typedef __bf16 v4bf  __attribute__((ext_vector_type(4)));
typedef float  v8f   __attribute__((ext_vector_type(8)));

// ---------------- problem dims ----------------
#define NNODES 25000
#define NE     100000
#define EP     (NE + NNODES)      // edges + self loops = 125000
#define NG     512
#define HEADS  10
#define FXD    78
#define HF     (HEADS * FXD)      // 780
#define SEQ    1000
#define EMBD   128
#define NFILT  32
#define KCONV  8
#define CONVOUT 121
#define KIM2   (SEQ * KCONV)      // 8000
#define XTF    (NFILT * CONVOUT)  // 3872

// LDS tile row stride (bf16 elems): 40*2 = 80B -> 16B aligned, bank-padded
#define LDS_STRIDE 40

// order-preserving float <-> uint encoding for atomic segment-max
__device__ __forceinline__ unsigned fenc(float f) {
  unsigned u = __float_as_uint(f);
  return (u & 0x80000000u) ? ~u : (u | 0x80000000u);
}
__device__ __forceinline__ float fdec(unsigned k) {
  unsigned u = (k & 0x80000000u) ? (k & 0x7FFFFFFFu) : ~k;
  return __uint_as_float(u);
}

__device__ __forceinline__ void edge_sd(const int* __restrict__ ei, int e, int& s, int& d) {
  if (e < NE) { s = ei[e]; d = ei[NE + e]; }
  else        { s = e - NE; d = e - NE; }   // self loop
}

// 16-bf16 WMMA fragment from LDS: two aligned 16B reads (ds_load_b128)
__device__ __forceinline__ v16bf ldfrag(const __bf16* base) {
  v8bf lo = *(const v8bf*)(base);
  v8bf hi = *(const v8bf*)(base + 16);
  return __builtin_shufflevector(lo, hi, 0,1,2,3,4,5,6,7,8,9,10,11,12,13,14,15);
}

// =====================================================================
// Packed-bf16 WMMA GEMM: C[M,N] = Abf[Mp,Kp] @ BT[Np,Kp]^T (+bias,+relu)
// Operands are pre-padded bf16 (A row-major, B transposed K-contiguous):
// the K-loop has NO guards, NO conversions, NO divergent branches.
// 256-thread block (8 waves) -> 128x64 tile; wave -> 32x32 (2x2 WMMA).
// One-step register prefetch pipeline overlaps global loads with WMMA.
// =====================================================================
__global__ __launch_bounds__(256)
void wmma_gemm_bf(const __bf16* __restrict__ A, const __bf16* __restrict__ BT,
                  const float* __restrict__ bias, float* __restrict__ C,
                  int M, int N, int Kp, int ldc, int relu)
{
  __shared__ __bf16 As[128][LDS_STRIDE];   // [row][k]
  __shared__ __bf16 Bs[64][LDS_STRIDE];    // [col][k]

  const int lane = threadIdx.x & 31;
  const int wave = threadIdx.x >> 5;
  const int nbn = (N + 63) >> 6;
  const int bm = blockIdx.x / nbn;
  const int bn = blockIdx.x % nbn;
  const int wm = wave >> 1;                // 0..3 : 32-row strip
  const int wn = wave & 1;                 // 0..1 : 32-col strip
  const int l15 = lane & 15;
  const int kph = (lane >> 4) * 8;

  const int a_r  = threadIdx.x >> 1;              // 0..127
  const int a_ch = (threadIdx.x & 1) * 16;        // 0 / 16
  const __bf16* ag = A + (size_t)(bm * 128 + a_r) * Kp + a_ch;

  const int b_c  = threadIdx.x >> 2;              // 0..63
  const int b_kg = (threadIdx.x & 3) * 8;         // 0/8/16/24
  const __bf16* bg = BT + (size_t)(bn * 64 + b_c) * Kp + b_kg;

  v8f acc[2][2] = {{v8f{}, v8f{}}, {v8f{}, v8f{}}};

  v8bf a0 = *(const v8bf*)(ag);
  v8bf a1 = *(const v8bf*)(ag + 8);
  v8bf b0 = *(const v8bf*)(bg);

  for (int k0 = 0; k0 < Kp; k0 += 32) {
    *(v8bf*)&As[a_r][a_ch]     = a0;
    *(v8bf*)&As[a_r][a_ch + 8] = a1;
    *(v8bf*)&Bs[b_c][b_kg]     = b0;
    __syncthreads();

    const int kn = k0 + 32;
    if (kn < Kp) {                    // uniform branch: register prefetch
      a0 = *(const v8bf*)(ag + kn);
      a1 = *(const v8bf*)(ag + kn + 8);
      b0 = *(const v8bf*)(bg + kn);
    }

    v16bf afr0 = ldfrag(&As[wm * 32      + l15][kph]);
    v16bf afr1 = ldfrag(&As[wm * 32 + 16 + l15][kph]);
    v16bf bfr0 = ldfrag(&Bs[wn * 32      + l15][kph]);
    v16bf bfr1 = ldfrag(&Bs[wn * 32 + 16 + l15][kph]);
    acc[0][0] = __builtin_amdgcn_wmma_f32_16x16x32_bf16(false, afr0, false, bfr0, (short)0, acc[0][0], false, false);
    acc[0][1] = __builtin_amdgcn_wmma_f32_16x16x32_bf16(false, afr0, false, bfr1, (short)0, acc[0][1], false, false);
    acc[1][0] = __builtin_amdgcn_wmma_f32_16x16x32_bf16(false, afr1, false, bfr0, (short)0, acc[1][0], false, false);
    acc[1][1] = __builtin_amdgcn_wmma_f32_16x16x32_bf16(false, afr1, false, bfr1, (short)0, acc[1][1], false, false);
    __syncthreads();
  }

  #pragma unroll
  for (int ms = 0; ms < 2; ++ms) {
    #pragma unroll
    for (int ns = 0; ns < 2; ++ns) {
      const int col = bn * 64 + wn * 32 + ns * 16 + l15;
      if (col >= N) continue;
      const float bv = bias ? bias[col] : 0.f;
      #pragma unroll
      for (int v = 0; v < 8; ++v) {
        const int m = bm * 128 + wm * 32 + ms * 16 + v + ((lane >> 4) << 3);
        if (m < M) {
          float r = acc[ms][ns][v] + bv;
          if (relu) r = fmaxf(r, 0.f);
          C[(size_t)m * ldc + col] = r;
        }
      }
    }
  }
}

// =====================================================================
// Conv1d branch as gathered WMMA GEMM (per output position j):
// D[g,o] = sum_k embbf[target[g,k/8]][j + k%8] * BcT[o][k].
// kb is 8-aligned -> each 8-run hits ONE token: 2 token loads + 16
// contiguous bf16 reads per fragment, branch-free. BcT tile in LDS.
// 3872 tiles = 484 blocks * 8 waves exactly (no divergent barriers).
// =====================================================================
__global__ __launch_bounds__(256)
void conv_wmma_gemm(const int* __restrict__ target, const __bf16* __restrict__ embbf,
                    const __bf16* __restrict__ BcT, const float* __restrict__ conv_b,
                    float* __restrict__ xtbuf)
{
  __shared__ __bf16 Bs[NFILT][LDS_STRIDE];   // [col][k]

  const int lane = threadIdx.x & 31;
  const int wave = threadIdx.x >> 5;
  const int tile = blockIdx.x * 8 + wave;    // < 3872 by construction
  const int ntm  = NG / 16;                  // 32
  const int j  = tile / ntm;
  const int mt = tile % ntm;
  const int l15 = lane & 15;
  const int g   = mt * 16 + l15;
  const int kph = (lane >> 4) * 8;

  const int b_c  = threadIdx.x & 31;         // 0..31 cols
  const int b_kq = (threadIdx.x >> 5) * 4;   // 0..28 step 4
  const __bf16* bg = BcT + (size_t)b_c * KIM2 + b_kq;

  v8f acc[2] = {v8f{}, v8f{}};
  v4bf b0 = *(const v4bf*)(bg);

  for (int k0 = 0; k0 < KIM2; k0 += 32) {
    *(v4bf*)&Bs[b_c][b_kq] = b0;
    __syncthreads();

    const int kn = k0 + 32;
    if (kn < KIM2) b0 = *(const v4bf*)(bg + kn);

    // gathered A fragment: kb 8-aligned => one token per 8-run
    const int kb = k0 + kph;
    const int c0 = kb >> 3;
    const int tk0 = target[g * SEQ + c0];
    const int tk1 = target[g * SEQ + c0 + 2];
    const __bf16* e0 = embbf + tk0 * EMBD + j;
    const __bf16* e1 = embbf + tk1 * EMBD + j;
    v16bf a;
    #pragma unroll
    for (int i = 0; i < 8; ++i) { a[i] = e0[i]; a[i + 8] = e1[i]; }

    v16bf bfr0 = ldfrag(&Bs[l15][kph]);
    v16bf bfr1 = ldfrag(&Bs[16 + l15][kph]);
    acc[0] = __builtin_amdgcn_wmma_f32_16x16x32_bf16(false, a, false, bfr0, (short)0, acc[0], false, false);
    acc[1] = __builtin_amdgcn_wmma_f32_16x16x32_bf16(false, a, false, bfr1, (short)0, acc[1], false, false);
    __syncthreads();
  }

  #pragma unroll
  for (int s = 0; s < 2; ++s) {
    const int o = s * 16 + l15;
    const float bv = conv_b[o];
    #pragma unroll
    for (int v = 0; v < 8; ++v) {
      const int m = mt * 16 + v + ((lane >> 4) << 3);
      xtbuf[(size_t)m * XTF + o * CONVOUT + j] = acc[s][v] + bv;
    }
  }
}

// ---------------- pack kernels (run once per operand) ----------------
// A[M,K] f32 -> Abf[Mp,Kp] bf16, zero padded
__global__ void k_pack_a(const float* __restrict__ A, __bf16* __restrict__ out,
                         int M, int K, int lda, int Kp, int total)
{
  int i = blockIdx.x * blockDim.x + threadIdx.x;
  if (i >= total) return;
  int r = i / Kp, k = i % Kp;
  float v = (r < M && k < K) ? A[(size_t)r * lda + k] : 0.f;
  out[i] = (__bf16)v;
}
// B[K,N] f32 -> BT[Np,Kp] bf16 transposed, zero padded
__global__ void k_pack_bT(const float* __restrict__ B, __bf16* __restrict__ out,
                          int N, int K, int ldb, int Kp, int total)
{
  int i = blockIdx.x * blockDim.x + threadIdx.x;
  if (i >= total) return;
  int c = i / Kp, k = i % Kp;
  float v = (c < N && k < K) ? B[(size_t)k * ldb + c] : 0.f;
  out[i] = (__bf16)v;
}
// straight f32 -> bf16 convert
__global__ void k_cvt_bf(const float* __restrict__ src, __bf16* __restrict__ dst, int n)
{
  int i = blockIdx.x * blockDim.x + threadIdx.x;
  if (i < n) dst[i] = (__bf16)src[i];
}

// ---- GAT attention coefficients a_src/a_dst per (node, head) ----
__global__ void k_attn_coef(const float* __restrict__ h, const float* __restrict__ att_src,
                            const float* __restrict__ att_dst,
                            float* __restrict__ asrc, float* __restrict__ adst)
{
  int t = blockIdx.x * blockDim.x + threadIdx.x;
  if (t >= NNODES * HEADS) return;
  int n = t / HEADS, hd = t % HEADS;
  const float* hp = h + n * HF + hd * FXD;
  const float* ws = att_src + hd * FXD;
  const float* wd = att_dst + hd * FXD;
  float sa = 0.f, sd = 0.f;
  for (int f = 0; f < FXD; ++f) { sa += hp[f] * ws[f]; sd += hp[f] * wd[f]; }
  asrc[t] = sa; adst[t] = sd;
}

// ---- pass 1: leaky-relu logits, segment-max, degree ----
__global__ void k_edge_attn(const int* __restrict__ ei, const float* __restrict__ asrc,
                            const float* __restrict__ adst, float* __restrict__ aedge,
                            unsigned* __restrict__ amaxk, float* __restrict__ deg)
{
  int t = blockIdx.x * blockDim.x + threadIdx.x;
  if (t >= EP * HEADS) return;
  int e = t / HEADS, hd = t % HEADS;
  int s, d; edge_sd(ei, e, s, d);
  float a = asrc[s * HEADS + hd] + adst[d * HEADS + hd];
  a = (a >= 0.f) ? a : 0.2f * a;
  aedge[t] = a;
  atomicMax(&amaxk[d * HEADS + hd], fenc(a));
  if (hd == 0) atomicAdd(&deg[d], 1.0f);
}

// ---- pass 2: exp & segment-sum ----
__global__ void k_edge_exp(const int* __restrict__ ei, float* __restrict__ aedge,
                           const unsigned* __restrict__ amaxk, float* __restrict__ denom)
{
  int t = blockIdx.x * blockDim.x + threadIdx.x;
  if (t >= EP * HEADS) return;
  int e = t / HEADS, hd = t % HEADS;
  int s, d; edge_sd(ei, e, s, d);
  float v = __expf(aedge[t] - fdec(amaxk[d * HEADS + hd]));
  aedge[t] = v;
  atomicAdd(&denom[d * HEADS + hd], v);
}

// ---- pass 3: weighted message scatter (one wave per edge) ----
__global__ void k_edge_msg(const int* __restrict__ ei, const float* __restrict__ h,
                           const float* __restrict__ aedge, const float* __restrict__ denom,
                           float* __restrict__ x1)
{
  int e = blockIdx.x * (blockDim.x >> 5) + (threadIdx.x >> 5);
  if (e >= EP) return;
  int lane = threadIdx.x & 31;
  int s, d; edge_sd(ei, e, s, d);
  for (int idx = lane; idx < HF; idx += 32) {
    int hd = idx / FXD;
    float alpha = aedge[e * HEADS + hd] / (denom[d * HEADS + hd] + 1e-16f);
    atomicAdd(&x1[d * HF + idx], h[s * HF + idx] * alpha);
  }
}

// ---- GCN normalized scatter (one wave per edge) ----
__global__ void k_edge_gcn(const int* __restrict__ ei, const float* __restrict__ dinv,
                           const float* __restrict__ h2, float* __restrict__ x2)
{
  int e = blockIdx.x * (blockDim.x >> 5) + (threadIdx.x >> 5);
  if (e >= EP) return;
  int lane = threadIdx.x & 31;
  int s, d; edge_sd(ei, e, s, d);
  float nrm = dinv[s] * dinv[d];
  for (int idx = lane; idx < HF; idx += 32)
    atomicAdd(&x2[d * HF + idx], h2[s * HF + idx] * nrm);
}

__global__ void k_bias_relu(float* __restrict__ p, const float* __restrict__ b, int rows, int cols)
{
  int i = blockIdx.x * blockDim.x + threadIdx.x;
  if (i >= rows * cols) return;
  p[i] = fmaxf(p[i] + b[i % cols], 0.f);
}

__global__ void k_dinv(const float* __restrict__ deg, float* __restrict__ dinv)
{
  int n = blockIdx.x * blockDim.x + threadIdx.x;
  if (n >= NNODES) return;
  float d = deg[n];
  dinv[n] = (d > 0.f) ? rsqrtf(d) : 0.f;
}

__global__ void k_cnt(const int* __restrict__ batch, float* __restrict__ cnt)
{
  int n = blockIdx.x * blockDim.x + threadIdx.x;
  if (n >= NNODES) return;
  atomicAdd(&cnt[batch[n]], 1.0f);
}

__global__ void k_pool(const float* __restrict__ x2, const int* __restrict__ batch,
                       float* __restrict__ gsum, unsigned* __restrict__ gmaxk)
{
  int i = blockIdx.x * blockDim.x + threadIdx.x;
  if (i >= NNODES * HF) return;
  int n = i / HF, j = i % HF;
  int g = batch[n];
  float v = x2[i];
  atomicAdd(&gsum[g * HF + j], v);
  atomicMax(&gmaxk[g * HF + j], fenc(v));
}

__global__ void k_pool_fin(const float* __restrict__ gsum, const unsigned* __restrict__ gmaxk,
                           const float* __restrict__ cnt, float* __restrict__ xg)
{
  int i = blockIdx.x * blockDim.x + threadIdx.x;
  if (i >= NG * HF) return;
  int g = i / HF, j = i % HF;
  float c = cnt[g];
  float mean = gsum[i] / fmaxf(c, 1.0f);
  float mx = (c > 0.f) ? fdec(gmaxk[i]) : 0.f;
  xg[g * (2 * HF) + j]      = mx;
  xg[g * (2 * HF) + HF + j] = mean;
}

// =====================================================================
static inline size_t al256(size_t x) { return (x + 255) & ~(size_t)255; }

extern "C" void kernel_launch(void* const* d_in, const int* in_sizes, int n_in,
                              void* d_out, int out_size, void* d_ws, size_t ws_size,
                              hipStream_t stream)
{
  const float* x        = (const float*)d_in[0];
  const int*   ei       = (const int*)  d_in[1];
  const int*   batch    = (const int*)  d_in[2];
  const int*   target   = (const int*)  d_in[3];
  const float* gat_W    = (const float*)d_in[4];
  const float* att_src  = (const float*)d_in[5];
  const float* att_dst  = (const float*)d_in[6];
  const float* gat_b    = (const float*)d_in[7];
  const float* gcn_W    = (const float*)d_in[8];
  const float* gcn_b    = (const float*)d_in[9];
  const float* fcg1_W   = (const float*)d_in[10];
  const float* fcg1_b   = (const float*)d_in[11];
  const float* fcg2_W   = (const float*)d_in[12];
  const float* fcg2_b   = (const float*)d_in[13];
  const float* emb      = (const float*)d_in[14];
  const float* conv_W   = (const float*)d_in[15];
  const float* conv_b   = (const float*)d_in[16];
  const float* fcxt_W   = (const float*)d_in[17];
  const float* fcxt_b   = (const float*)d_in[18];
  const float* fc1_W    = (const float*)d_in[19];
  const float* fc1_b    = (const float*)d_in[20];
  const float* fc2_W    = (const float*)d_in[21];
  const float* fc2_b    = (const float*)d_in[22];
  const float* out_W    = (const float*)d_in[23];
  const float* out_b    = (const float*)d_in[24];
  float* out = (float*)d_out;

  // ---- workspace carve-up ----
  char* w = (char*)d_ws;
  size_t off = 0;
  auto carve = [&](size_t bytes) { void* p = w + off; off = al256(off + bytes); return p; };
  float*    hbuf  = (float*)   carve((size_t)NNODES * HF * 4);   // GAT features; reused as x2
  float*    x1    = (float*)   carve((size_t)NNODES * HF * 4);
  float*    h2    = (float*)   carve((size_t)NNODES * HF * 4);
  float*    x2    = hbuf;
  float*    asrc  = (float*)   carve((size_t)NNODES * HEADS * 4);
  float*    adst  = (float*)   carve((size_t)NNODES * HEADS * 4);
  float*    aedge = (float*)   carve((size_t)EP * HEADS * 4);
  unsigned* amaxk = (unsigned*)carve((size_t)NNODES * HEADS * 4);
  float*    denom = (float*)   carve((size_t)NNODES * HEADS * 4);
  float*    deg   = (float*)   carve((size_t)NNODES * 4);
  float*    dinv  = (float*)   carve((size_t)NNODES * 4);
  float*    cnt   = (float*)   carve((size_t)NG * 4);
  float*    gsum  = (float*)   carve((size_t)NG * HF * 4);
  unsigned* gmaxk = (unsigned*)carve((size_t)NG * HF * 4);
  float*    xg    = (float*)   carve((size_t)NG * 2 * HF * 4);
  float*    t1    = (float*)   carve((size_t)NG * 1024 * 4);
  float*    t2    = (float*)   carve((size_t)NG * 512 * 4);
  float*    xc    = (float*)   carve((size_t)NG * 256 * 4);
  float*    xtbuf = (float*)   carve((size_t)NG * XTF * 4);
  // bf16 packed operands
  __bf16* xbf   = (__bf16*)carve((size_t)25088 * 96 * 2);
  __bf16* w1T   = (__bf16*)carve((size_t)832 * 96 * 2);
  __bf16* x1bf  = (__bf16*)carve((size_t)25088 * 800 * 2);
  __bf16* w2T   = (__bf16*)carve((size_t)832 * 800 * 2);
  __bf16* xgbf  = (__bf16*)carve((size_t)512 * 1568 * 2);
  __bf16* w3T   = (__bf16*)carve((size_t)1024 * 1568 * 2);
  __bf16* t1bf  = (__bf16*)carve((size_t)512 * 1024 * 2);
  __bf16* w4T   = (__bf16*)carve((size_t)128 * 1024 * 2);
  __bf16* embbf = (__bf16*)carve((size_t)26 * EMBD * 2);
  __bf16* BcT   = (__bf16*)carve((size_t)NFILT * KIM2 * 2);
  __bf16* xtbf  = (__bf16*)carve((size_t)512 * 3872 * 2);
  __bf16* w5T   = (__bf16*)carve((size_t)128 * 3872 * 2);
  __bf16* xcbf  = (__bf16*)carve((size_t)512 * 256 * 2);
  __bf16* w6T   = (__bf16*)carve((size_t)1024 * 256 * 2);
  __bf16* w7T   = (__bf16*)carve((size_t)512 * 1024 * 2);
  __bf16* t2bf  = (__bf16*)carve((size_t)512 * 512 * 2);
  __bf16* w8T   = (__bf16*)carve((size_t)64 * 512 * 2);
  (void)ws_size; (void)n_in; (void)in_sizes; (void)out_size;

  auto g1d = [](long long n) { return (int)((n + 255) / 256); };
  auto packA = [&](const float* A, __bf16* dst, int M, int K, int lda) {
    int Mp = ((M + 127) / 128) * 128, Kp = ((K + 31) / 32) * 32;
    int total = Mp * Kp;
    k_pack_a<<<g1d(total), 256, 0, stream>>>(A, dst, M, K, lda, Kp, total);
  };
  auto packBT = [&](const float* B, __bf16* dst, int N, int K, int ldb) {
    int Np = ((N + 63) / 64) * 64, Kp = ((K + 31) / 32) * 32;
    int total = Np * Kp;
    k_pack_bT<<<g1d(total), 256, 0, stream>>>(B, dst, N, K, ldb, Kp, total);
  };
  auto gemm = [&](const __bf16* Abf, const __bf16* BT, const float* bias, float* C,
                  int M, int N, int Kp, int ldc, int relu) {
    int blocks = ((M + 127) / 128) * ((N + 63) / 64);
    wmma_gemm_bf<<<blocks, 256, 0, stream>>>(Abf, BT, bias, C, M, N, Kp, ldc, relu);
  };

  // ---- 0. pack all weights (independent of activations) ----
  packBT(gat_W,  w1T, HF,   FXD,    HF);     // Np 832,  Kp 96
  packBT(gcn_W,  w2T, HF,   HF,     HF);     // Np 832,  Kp 800
  packBT(fcg1_W, w3T, 1024, 2 * HF, 1024);   // Np 1024, Kp 1568
  packBT(fcg2_W, w4T, 128,  1024,   128);
  packBT(fcxt_W, w5T, 128,  XTF,    128);
  packBT(fc1_W,  w6T, 1024, 256,    1024);
  packBT(fc2_W,  w7T, 512,  1024,   512);
  packBT(out_W,  w8T, 1,    512,    1);      // Np 64
  k_cvt_bf<<<g1d(26 * EMBD), 256, 0, stream>>>(emb, embbf, 26 * EMBD);
  k_cvt_bf<<<g1d(NFILT * KIM2), 256, 0, stream>>>(conv_W, BcT, NFILT * KIM2);  // (O,C,T) is already K-major

  // ---- 1. GAT linear: h = x @ gat_W ----
  packA(x, xbf, NNODES, FXD, FXD);
  gemm(xbf, w1T, nullptr, hbuf, NNODES, HF, 96, HF, 0);

  // ---- 2. attention coefficients ----
  k_attn_coef<<<g1d((long long)NNODES * HEADS), 256, 0, stream>>>(hbuf, att_src, att_dst, asrc, adst);

  // ---- 3. zero accumulators ----
  hipMemsetAsync(amaxk, 0, (size_t)NNODES * HEADS * 4, stream);
  hipMemsetAsync(denom, 0, (size_t)NNODES * HEADS * 4, stream);
  hipMemsetAsync(deg,   0, (size_t)NNODES * 4, stream);
  hipMemsetAsync(x1,    0, (size_t)NNODES * HF * 4, stream);

  // ---- 4-6. GAT softmax + message passing ----
  k_edge_attn<<<g1d((long long)EP * HEADS), 256, 0, stream>>>(ei, asrc, adst, aedge, amaxk, deg);
  k_edge_exp <<<g1d((long long)EP * HEADS), 256, 0, stream>>>(ei, aedge, amaxk, denom);
  k_edge_msg <<<(EP + 7) / 8, 256, 0, stream>>>(ei, hbuf, aedge, denom, x1);
  k_bias_relu<<<g1d((long long)NNODES * HF), 256, 0, stream>>>(x1, gat_b, NNODES, HF);

  // ---- 7-8. GCN ----
  k_dinv<<<g1d(NNODES), 256, 0, stream>>>(deg, dinv);
  packA(x1, x1bf, NNODES, HF, HF);
  gemm(x1bf, w2T, nullptr, h2, NNODES, HF, 800, HF, 0);
  hipMemsetAsync(x2, 0, (size_t)NNODES * HF * 4, stream);      // hbuf is dead now
  k_edge_gcn<<<(EP + 7) / 8, 256, 0, stream>>>(ei, dinv, h2, x2);
  k_bias_relu<<<g1d((long long)NNODES * HF), 256, 0, stream>>>(x2, gcn_b, NNODES, HF);

  // ---- 9. graph pooling (max | mean) ----
  hipMemsetAsync(gsum,  0, (size_t)NG * HF * 4, stream);
  hipMemsetAsync(gmaxk, 0, (size_t)NG * HF * 4, stream);
  hipMemsetAsync(cnt,   0, (size_t)NG * 4, stream);
  k_cnt <<<g1d(NNODES), 256, 0, stream>>>(batch, cnt);
  k_pool<<<g1d((long long)NNODES * HF), 256, 0, stream>>>(x2, batch, gsum, gmaxk);
  k_pool_fin<<<g1d((long long)NG * HF), 256, 0, stream>>>(gsum, gmaxk, cnt, xg);

  // ---- 10. graph MLP head -> xc[:, 0:128] ----
  packA(xg, xgbf, NG, 2 * HF, 2 * HF);
  gemm(xgbf, w3T, fcg1_b, t1, NG, 1024, 1568, 1024, 1);
  packA(t1, t1bf, NG, 1024, 1024);
  gemm(t1bf, w4T, fcg2_b, xc, NG, 128, 1024, 256, 0);

  // ---- 11. target CNN branch -> xc[:, 128:256] ----
  conv_wmma_gemm<<<(CONVOUT * (NG / 16)) / 8, 256, 0, stream>>>(target, embbf, BcT, conv_b, xtbuf);
  packA(xtbuf, xtbf, NG, XTF, XTF);
  gemm(xtbf, w5T, fcxt_b, xc + 128, NG, 128, 3872, 256, 0);

  // ---- 12. fusion head ----
  packA(xc, xcbf, NG, 256, 256);
  gemm(xcbf, w6T, fc1_b, t1, NG, 1024, 256, 1024, 1);
  packA(t1, t1bf, NG, 1024, 1024);
  gemm(t1bf, w7T, fc2_b, t2, NG, 512, 1024, 512, 1);
  packA(t2, t2bf, NG, 512, 512);
  gemm(t2bf, w8T, out_b, out, NG, 1, 512, 1, 0);
}
